// WindowAttention_36610301231822
// MI455X (gfx1250) — matrix operand order repaired and checked by reading, not compiled, not run on hardware
//
#include <hip/hip_runtime.h>

// ---------------------------------------------------------------------------
// Fused Swin window-attention forward for MI455X (gfx1250, wave32, WMMA).
//
// prep_kernel: pads/converts weights to bf16 in d_ws (zero-padded so the main
//   kernel has branch-free contiguous fragment loads), pre-scales the q slice
//   by 30^-0.5, and pre-gathers the relative-position bias to [H][64][64].
//
// win_attn_fused: one workgroup (256 thr = 8 wave32) per window.
//   qkv GEMM -> (q|k in LDS, v transposed in LDS) -> S=qk^T +bias+mask ->
//   softmax -> P@V -> proj GEMM -> fp32 out.
// HBM traffic ~ x-in + out (~380 MB) -> ~16us floor at 23.3 TB/s; weights and
// bias stay in the 192 MB L2. All matmuls: v_wmma_f32_16x16x32_bf16.
// ---------------------------------------------------------------------------

typedef __bf16 bf16;
typedef __attribute__((ext_vector_type(16))) __bf16 v16bf;
typedef __attribute__((ext_vector_type(8)))  __bf16 v8bf;
typedef __attribute__((ext_vector_type(8)))  float  v8f;

union v16u { v16bf v; v8bf h[2]; };

#define DIMC   180
#define NH     6
#define HD     30
#define NT     64          // tokens per window
#define KPAD   192         // 180 padded to 6*32
#define QKCOLS 384         // q|k padded: 2 * 6 heads * 32
#define WQROWS 576         // 3 * 6 heads * 32
#define TABLE_SZ 225
#define NWMASK 63

// d_ws byte offsets (all 256-aligned)
#define WS_WQ    0               // bf16 [576][192]  = 221184 B
#define WS_WP    221184          // bf16 [192][192]  =  73728 B
#define WS_BIAS  294912          // f32  [6][64][64] =  98304 B
#define WS_BQ    393216          // f32  [576]       =   2304 B
#define WS_BP    395520          // f32  [192]       =    768 B
#define PREP_TOTAL (110592 + 36864 + 24576 + 576 + 192)   // 172800

__device__ __forceinline__ v8f wmma_bf16(v16bf a, v16bf b, v8f c) {
  return __builtin_amdgcn_wmma_f32_16x16x32_bf16(
      false, a, false, b, (short)0, c, false, false);
}

__device__ __forceinline__ v8f vzero8() {
  v8f z = {0.f, 0.f, 0.f, 0.f, 0.f, 0.f, 0.f, 0.f};
  return z;
}

__device__ __forceinline__ v16bf ld2x8(const bf16* p0, const bf16* p1) {
  v16u r;
  r.h[0] = *(const v8bf*)p0;
  r.h[1] = *(const v8bf*)p1;
  return r.v;
}

// ---------------------------------------------------------------------------
__global__ __launch_bounds__(256)
void prep_kernel(const float* __restrict__ w_qkv, const float* __restrict__ b_qkv,
                 const float* __restrict__ rpb,   const int*   __restrict__ rpi,
                 const float* __restrict__ w_proj,const float* __restrict__ b_proj,
                 bf16* __restrict__ wq, bf16* __restrict__ wp,
                 float* __restrict__ biasw, float* __restrict__ bq,
                 float* __restrict__ bp)
{
  const float scale = 0.18257418583505536f;   // 30^-0.5
  int i = blockIdx.x * 256 + threadIdx.x;
  if (i < WQROWS * KPAD) {                    // padded qkv weight, q pre-scaled
    int col = i / KPAD, c = i - col * KPAD;
    int which = col / 192, rem = col - which * 192;
    int h = rem >> 5, d = rem & 31;
    float v = 0.f;
    if (d < HD && c < DIMC) v = w_qkv[(which * DIMC + h * HD + d) * DIMC + c];
    if (which == 0) v *= scale;
    wq[i] = (bf16)v;
    return;
  }
  i -= WQROWS * KPAD;
  if (i < KPAD * KPAD) {                      // padded proj weight
    int o = i / KPAD, c = i - o * KPAD;
    float v = (o < DIMC && c < DIMC) ? w_proj[o * DIMC + c] : 0.f;
    wp[i] = (bf16)v;
    return;
  }
  i -= KPAD * KPAD;
  if (i < NH * NT * NT) {                     // gathered relative-position bias
    int h = i >> 12, mn = i & 4095;
    int t = rpi[mn];
    t = t < 0 ? 0 : (t >= TABLE_SZ ? TABLE_SZ - 1 : t);
    biasw[i] = rpb[t * NH + h];
    return;
  }
  i -= NH * NT * NT;
  if (i < WQROWS) {                           // padded qkv bias, q pre-scaled
    int which = i / 192, rem = i - which * 192;
    int h = rem >> 5, d = rem & 31;
    float v = (d < HD) ? b_qkv[which * DIMC + h * HD + d] : 0.f;
    if (which == 0) v *= scale;
    bq[i] = v;
    return;
  }
  i -= WQROWS;
  if (i < KPAD) bp[i] = (i < DIMC) ? b_proj[i] : 0.f;
}

// ---------------------------------------------------------------------------
__global__ __launch_bounds__(256)
void win_attn_fused(const float* __restrict__ x,
                    const float* __restrict__ mask,
                    const bf16*  __restrict__ wq,
                    const bf16*  __restrict__ wp,
                    const float* __restrict__ biasw,
                    const float* __restrict__ bq,
                    const float* __restrict__ bp,
                    float* __restrict__ out)
{
  __shared__ bf16 sX[NT][KPAD];        // 24 KB  x tile (bf16, zero-padded K)
  __shared__ bf16 sQK[NT][QKCOLS];     // 48 KB  q|k, head-dim padded to 32
  __shared__ bf16 sVT[NH][32][NT];     // 24 KB  v transposed: [h][d][token]
  __shared__ bf16 sP[NH][16][NT];      // 12 KB  softmax band per head
  __shared__ bf16 sO[NT][KPAD];        // 24 KB  attention output (padded)

  const int tid  = threadIdx.x;
  const int lane = tid & 31;
  const int wave = tid >> 5;
  const int g    = lane >> 4;          // half-wave group (0/1)
  const int ln   = lane & 15;
  const int b    = blockIdx.x;

  // ---------------- phase 0: stage x -> LDS (bf16), zero pads --------------
  const float* xb = x + (size_t)b * NT * DIMC;
  for (int i = tid; i < NT * DIMC; i += 256) {
    int n = i / DIMC, c = i - n * DIMC;
    sX[n][c] = (bf16)xb[i];
  }
  for (int i = tid; i < NT * (KPAD - DIMC); i += 256) {
    int n = i / (KPAD - DIMC), c = DIMC + (i - n * (KPAD - DIMC));
    sX[n][c] = (bf16)0.f;
    sO[n][c] = (bf16)0.f;
  }
  __syncthreads();

  // ---------------- phase 1: QKV = X @ Wq_pad^T + bq (padded col space) ----
  for (int nt = wave; nt < WQROWS / 16; nt += 8) {
    const int col = nt * 16 + ln;                  // 0..575
    const bf16* wrow = wq + col * KPAD;
    v8f acc[4];
    #pragma unroll
    for (int mt = 0; mt < 4; ++mt) acc[mt] = vzero8();
    #pragma unroll
    for (int kt = 0; kt < 6; ++kt) {
      v16bf bfrag = ld2x8(wrow + kt * 32 + g * 16,
                          wrow + kt * 32 + g * 16 + 8);
      #pragma unroll
      for (int mt = 0; mt < 4; ++mt) {
        int m = mt * 16 + ln;
        v16bf afrag = ld2x8(&sX[m][kt * 32 + g * 16],
                            &sX[m][kt * 32 + g * 16 + 8]);
        acc[mt] = wmma_bf16(afrag, bfrag, acc[mt]);
      }
    }
    const float bqv = bq[col];
    const int which = col / 192;
    const int rem = col - which * 192, h = rem >> 5, d = rem & 31;
    #pragma unroll
    for (int mt = 0; mt < 4; ++mt) {
      #pragma unroll
      for (int r = 0; r < 8; ++r) {
        int m = mt * 16 + g * 8 + r;
        bf16 v = (bf16)(acc[mt][r] + bqv);
        if (which < 2) sQK[m][col] = v;            // q (scaled) | k
        else           sVT[h][d][m] = v;           // v transposed
      }
    }
  }
  __syncthreads();

  // ---------------- phase 2: attention, one head per wave ------------------
  if (wave < NH) {
    const int h = wave;
    const bf16* qrow0 = &sQK[0][h * 32];
    const int kbase = 192 + h * 32;
    const float* maskw = mask + (size_t)(b & NWMASK) * NT * NT;
    const float* biash = biasw + h * NT * NT;

    v16bf bk[4];                                   // K^T: col=token j, k=d
    #pragma unroll
    for (int nt = 0; nt < 4; ++nt) {
      int j = nt * 16 + ln;
      bk[nt] = ld2x8(&sQK[j][kbase + g * 16], &sQK[j][kbase + g * 16 + 8]);
    }
    v16bf bv[2][2];                                // V: col=d, k=token j
    #pragma unroll
    for (int kt = 0; kt < 2; ++kt) {
      #pragma unroll
      for (int ot = 0; ot < 2; ++ot) {
        int d = ot * 16 + ln;                      // rows 30/31 are zero pads
        bv[kt][ot] = ld2x8(&sVT[h][d][kt * 32 + g * 16],
                           &sVT[h][d][kt * 32 + g * 16 + 8]);
      }
    }

    for (int mt = 0; mt < 4; ++mt) {
      int m = mt * 16 + ln;
      v16bf aq = ld2x8(&sQK[m][h * 32 + g * 16],
                       &sQK[m][h * 32 + g * 16 + 8]);
      v8f s[4];
      #pragma unroll
      for (int nt = 0; nt < 4; ++nt) s[nt] = wmma_bf16(aq, bk[nt], vzero8());

      #pragma unroll
      for (int nt = 0; nt < 4; ++nt) {             // + bias + mask (fp32)
        int n = nt * 16 + ln;
        #pragma unroll
        for (int r = 0; r < 8; ++r) {
          int mm = mt * 16 + g * 8 + r;
          s[nt][r] += biash[mm * NT + n] + maskw[mm * NT + n];
        }
      }

      #pragma unroll
      for (int r = 0; r < 8; ++r) {                // row softmax (16-lane grp)
        float mx = fmaxf(fmaxf(s[0][r], s[1][r]), fmaxf(s[2][r], s[3][r]));
        #pragma unroll
        for (int off = 8; off >= 1; off >>= 1)
          mx = fmaxf(mx, __shfl_xor(mx, off, 32));
        float sum = 0.f;
        #pragma unroll
        for (int nt = 0; nt < 4; ++nt) {
          float ev = __expf(s[nt][r] - mx);
          s[nt][r] = ev;
          sum += ev;
        }
        #pragma unroll
        for (int off = 8; off >= 1; off >>= 1)
          sum += __shfl_xor(sum, off, 32);
        float inv = 1.f / sum;
        #pragma unroll
        for (int nt = 0; nt < 4; ++nt) s[nt][r] *= inv;
      }

      #pragma unroll
      for (int nt = 0; nt < 4; ++nt) {             // P band -> LDS (bf16)
        int n = nt * 16 + ln;
        #pragma unroll
        for (int r = 0; r < 8; ++r)
          sP[h][g * 8 + r][n] = (bf16)s[nt][r];
      }

      v8f oacc[2] = { vzero8(), vzero8() };        // O band = P @ V
      #pragma unroll
      for (int kt = 0; kt < 2; ++kt) {
        v16bf ap = ld2x8(&sP[h][ln][kt * 32 + g * 16],
                         &sP[h][ln][kt * 32 + g * 16 + 8]);
        oacc[0] = wmma_bf16(ap, bv[kt][0], oacc[0]);
        oacc[1] = wmma_bf16(ap, bv[kt][1], oacc[1]);
      }
      #pragma unroll
      for (int ot = 0; ot < 2; ++ot) {
        int d = ot * 16 + ln;
        if (d < HD) {
          #pragma unroll
          for (int r = 0; r < 8; ++r) {
            int mm = mt * 16 + g * 8 + r;
            sO[mm][h * HD + d] = (bf16)oacc[ot][r];
          }
        }
      }
    }
  }
  __syncthreads();

  // ---------------- phase 3: out = O @ Wp_pad^T + bp -----------------------
  for (int nt = wave; nt < KPAD / 16; nt += 8) {
    const int o = nt * 16 + ln;                    // 0..191 (rows>=180 zero)
    const bf16* wrow = wp + o * KPAD;
    v8f acc[4];
    #pragma unroll
    for (int mt = 0; mt < 4; ++mt) acc[mt] = vzero8();
    #pragma unroll
    for (int kt = 0; kt < 6; ++kt) {
      v16bf bfrag = ld2x8(wrow + kt * 32 + g * 16,
                          wrow + kt * 32 + g * 16 + 8);
      #pragma unroll
      for (int mt = 0; mt < 4; ++mt) {
        int m = mt * 16 + ln;
        v16bf afrag = ld2x8(&sO[m][kt * 32 + g * 16],
                            &sO[m][kt * 32 + g * 16 + 8]);
        acc[mt] = wmma_bf16(afrag, bfrag, acc[mt]);
      }
    }
    if (o < DIMC) {
      float bpv = bp[o];
      float* ob = out + (size_t)b * NT * DIMC;
      #pragma unroll
      for (int mt = 0; mt < 4; ++mt) {
        #pragma unroll
        for (int r = 0; r < 8; ++r) {
          int m = mt * 16 + g * 8 + r;
          ob[m * DIMC + o] = acc[mt][r] + bpv;
        }
      }
    }
  }
}

// ---------------------------------------------------------------------------
extern "C" void kernel_launch(void* const* d_in, const int* in_sizes, int n_in,
                              void* d_out, int out_size, void* d_ws, size_t ws_size,
                              hipStream_t stream) {
  const float* x      = (const float*)d_in[0];
  const int*   rpi    = (const int*)  d_in[1];
  const float* mask   = (const float*)d_in[2];
  const float* w_qkv  = (const float*)d_in[3];
  const float* b_qkv  = (const float*)d_in[4];
  const float* rpb    = (const float*)d_in[5];
  const float* w_proj = (const float*)d_in[6];
  const float* b_proj = (const float*)d_in[7];
  float* outp = (float*)d_out;

  char* ws = (char*)d_ws;
  bf16*  wq    = (bf16*) (ws + WS_WQ);
  bf16*  wpp   = (bf16*) (ws + WS_WP);
  float* biasw = (float*)(ws + WS_BIAS);
  float* bq    = (float*)(ws + WS_BQ);
  float* bp    = (float*)(ws + WS_BP);

  hipLaunchKernelGGL(prep_kernel, dim3((PREP_TOTAL + 255) / 256), dim3(256),
                     0, stream,
                     w_qkv, b_qkv, rpb, rpi, w_proj, b_proj,
                     wq, wpp, biasw, bq, bp);

  const int n_windows = in_sizes[0] / (NT * DIMC);   // 4096
  hipLaunchKernelGGL(win_attn_fused, dim3(n_windows), dim3(256), 0, stream,
                     x, mask, wq, wpp, biasw, bq, bp, outp);
  (void)n_in; (void)out_size; (void)ws_size;
}
